// BiholoModelFuncGENERALforHYMinv3_78323023610571
// MI455X (gfx1250) — compile-verified, matrix operand order checked by reading, not run on hardware
//
#include <hip/hip_runtime.h>
#include <hip/hip_bf16.h>

// MI455X / gfx1250, wave32. bf16 WMMA path: v_wmma_f32_16x16x32_bf16.

typedef __attribute__((ext_vector_type(16))) __bf16 v16bf;
typedef __attribute__((ext_vector_type(8)))  __bf16 v8bf;
typedef __attribute__((ext_vector_type(8)))  float  v8f;

#define KFPAD 736    // 729 features padded to 23*32
#define HID   1024
#define PW    16     // partial width = HID/64 (grid.y of GEMM2)

// ---------------------------------------------------------------------------
// Feature decode helpers: second-level values with embedded signs.
// R2 (45): [r1 (x) r2 (36), -(i1 (x) i2) (9)]
// I2 (36): [r1 (x) i2 (18), -(i1 (x) r2) (18)]
// ---------------------------------------------------------------------------
__device__ __forceinline__ float lvl2_R(const float* r1, const float* i1,
                                        const float* r2, const float* i2, int a) {
    if (a < 36) return r1[a / 6] * r2[a % 6];
    int b = a - 36;
    return -(i1[b / 3] * i2[b % 3]);
}
__device__ __forceinline__ float lvl2_I(const float* r1, const float* i1,
                                        const float* r2, const float* i2, int a) {
    if (a < 18) return r1[a / 3] * i2[a % 3];
    int b = a - 18;
    return -(i1[b / 6] * r2[b % 6]);
}

// ---------------------------------------------------------------------------
// Kernel 1: biholomorphic section features -> bf16 [B, KFPAD]
// 256 threads = 8 rows x 32 lanes; factor tables live in LDS per row.
// ---------------------------------------------------------------------------
__global__ __launch_bounds__(256)
void feats_kernel(const float* __restrict__ x, __hip_bfloat16* __restrict__ feats, int B) {
    __shared__ float xsh[8][18];
    __shared__ float rsh[8][18];  // 3 blocks * 6 real factors
    __shared__ float ish[8][9];   // 3 blocks * 3 imag factors
    __shared__ float kinv[8];

    const int tid = threadIdx.x;
    const int rowBase = blockIdx.x * 8;
    if (tid < 144) {
        int rl = tid / 18, c = tid % 18;
        int rg = rowBase + rl;
        xsh[rl][c] = (rg < B) ? x[(size_t)rg * 18 + c] : 0.f;
    }
    __syncthreads();

    const int rl = tid >> 5;
    const int lane = tid & 31;
    if (lane == 0) {
        float kap = 1.f;
        #pragma unroll
        for (int blk = 0; blk < 3; ++blk) {
            float X0 = xsh[rl][3 * blk + 0], X1 = xsh[rl][3 * blk + 1], X2 = xsh[rl][3 * blk + 2];
            float Y0 = xsh[rl][9 + 3 * blk + 0], Y1 = xsh[rl][9 + 3 * blk + 1], Y2 = xsh[rl][9 + 3 * blk + 2];
            float r0 = X0 * X0 + Y0 * Y0, r1v = X0 * X1 + Y0 * Y1, r2v = X0 * X2 + Y0 * Y2;
            float r3v = X1 * X1 + Y1 * Y1, r4v = X1 * X2 + Y1 * Y2, r5v = X2 * X2 + Y2 * Y2;
            rsh[rl][6 * blk + 0] = r0;  rsh[rl][6 * blk + 1] = r1v; rsh[rl][6 * blk + 2] = r2v;
            rsh[rl][6 * blk + 3] = r3v; rsh[rl][6 * blk + 4] = r4v; rsh[rl][6 * blk + 5] = r5v;
            ish[rl][3 * blk + 0] = X0 * Y1 - X1 * Y0;
            ish[rl][3 * blk + 1] = X0 * Y2 - X2 * Y0;
            ish[rl][3 * blk + 2] = X1 * Y2 - X2 * Y1;
            kap *= (r0 + r3v + r5v);
        }
        kinv[rl] = 1.f / kap;
    }
    __syncthreads();

    const int rg = rowBase + rl;
    const float* r1 = &rsh[rl][0]; const float* r2 = r1 + 6; const float* r3 = r1 + 12;
    const float* i1 = &ish[rl][0]; const float* i2 = i1 + 3; const float* i3 = i1 + 6;
    const float inv = kinv[rl];
    __hip_bfloat16* orow = feats + (size_t)rg * KFPAD;

    #pragma unroll 1
    for (int j = 0; j < 23; ++j) {
        int f = lane + 32 * j;
        float v = 0.f;
        if (f < 729) {
            if (f < 378) {                                  // real part (378)
                if (f < 270)      v =  lvl2_R(r1, i1, r2, i2, f / 6)  * r3[f % 6];
                else { int g = f - 270;
                                  v = -lvl2_I(r1, i1, r2, i2, g / 3)  * i3[g % 3]; }
            } else {                                        // imag part (351)
                int f2 = f - 378;
                if (f2 < 135)     v =  lvl2_R(r1, i1, r2, i2, f2 / 3) * i3[f2 % 3];
                else { int g = f2 - 135;
                                  v = -lvl2_I(r1, i1, r2, i2, g / 6)  * r3[g % 6]; }
            }
            v *= inv;
        }
        if (rg < B) orow[f] = __float2bfloat16(v);
    }
}

// ---------------------------------------------------------------------------
// Kernel 2: weight prep  W[K,N] f32 (row-major) -> Wt[N,Kpad] bf16, zero-pad K.
// ---------------------------------------------------------------------------
__global__ __launch_bounds__(256)
void wprep_kernel(const float* __restrict__ W, __hip_bfloat16* __restrict__ Wt,
                  int K, int N, int Kpad) {
    int idx = blockIdx.x * blockDim.x + threadIdx.x;   // idx = k*N + n (coalesced reads)
    if (idx >= Kpad * N) return;
    int k = idx / N, n = idx % N;
    float v = (k < K) ? W[(size_t)k * N + n] : 0.f;
    Wt[(size_t)n * Kpad + k] = __float2bfloat16(v);
}

// ---------------------------------------------------------------------------
// Fragment load: 16-bit 16x32 striping (ISA 7.12.2).
//   lane l (half = l>>4): elems 0..7  = K(k0 + 8*half .. +7)
//                         elems 8..15 = K(k0 + 16 + 8*half .. +7)
// ---------------------------------------------------------------------------
__device__ __forceinline__ v16bf load_frag(const __bf16* __restrict__ p) {
    v8bf lo = *(const v8bf*)p;
    v8bf hi = *(const v8bf*)(p + 16);
    return __builtin_shufflevector(lo, hi,
             0, 1, 2, 3, 4, 5, 6, 7, 8, 9, 10, 11, 12, 13, 14, 15);
}

#define WMMA_BF16(A_, B_, C_) \
    __builtin_amdgcn_wmma_f32_16x16x32_bf16(false, (A_), false, (B_), (short)0, (C_), false, false)

// ---------------------------------------------------------------------------
// WMMA mainloop: one wave computes a 32x64 C tile (2 M-tiles x 4 N-tiles,
// 8 accumulators). K is a compile-time constant and the loop fully unrolls:
// all loads are immediate offsets off 3 base pointers (ap0, ap1, bp) -- no
// per-step address arithmetic, no buffer-rotation copies, so the scheduler
// can hoist loads across steps and issue WMMAs behind partial loadcnt waits.
// A row-major [M,K] bf16, Bt column-major-as-rows [N,K] bf16, K % 32 == 0.
// ---------------------------------------------------------------------------
template<int K>
__device__ __forceinline__ void gemm_tile(const __hip_bfloat16* __restrict__ A,
                                          const __hip_bfloat16* __restrict__ Bt,
                                          int M, int m0, int n0, int lane,
                                          v8f acc[2][4]) {
    const int half = lane >> 4;
    const int l15  = lane & 15;
    int ar0 = m0 + l15;      if (ar0 >= M) ar0 = M - 1;
    int ar1 = m0 + 16 + l15; if (ar1 >= M) ar1 = M - 1;
    const __bf16* ap0 = (const __bf16*)A  + (size_t)ar0 * K + half * 8;
    const __bf16* ap1 = (const __bf16*)A  + (size_t)ar1 * K + half * 8;
    const __bf16* bp  = (const __bf16*)Bt + (size_t)(n0 + l15) * K + half * 8;

    #pragma unroll
    for (int k0 = 0; k0 < K; k0 += 32) {
        v16bf a0 = load_frag(ap0 + k0);
        v16bf a1 = load_frag(ap1 + k0);
        v16bf b0 = load_frag(bp + k0);
        v16bf b1 = load_frag(bp + 16 * K + k0);
        v16bf b2 = load_frag(bp + 32 * K + k0);
        v16bf b3 = load_frag(bp + 48 * K + k0);

        acc[0][0] = WMMA_BF16(a0, b0, acc[0][0]);
        acc[1][0] = WMMA_BF16(a1, b0, acc[1][0]);
        acc[0][1] = WMMA_BF16(a0, b1, acc[0][1]);
        acc[1][1] = WMMA_BF16(a1, b1, acc[1][1]);
        acc[0][2] = WMMA_BF16(a0, b2, acc[0][2]);
        acc[1][2] = WMMA_BF16(a1, b2, acc[1][2]);
        acc[0][3] = WMMA_BF16(a0, b3, acc[0][3]);
        acc[1][3] = WMMA_BF16(a1, b3, acc[1][3]);
    }
}

// ---------------------------------------------------------------------------
// Kernel 3: GEMM layer 1:  C = (A @ W + b)^2  -> bf16 [M, N]
// block = 8 waves = 256 rows x 64 cols; grid = (ceil(M/256), N/64)
// ---------------------------------------------------------------------------
__global__ __launch_bounds__(256)
void gemm_sq_kernel(const __hip_bfloat16* __restrict__ A,
                    const __hip_bfloat16* __restrict__ Bt,
                    const float* __restrict__ bias,
                    __hip_bfloat16* __restrict__ C,
                    int M, int N) {
    const int wave = threadIdx.x >> 5, lane = threadIdx.x & 31;
    const int m0 = (blockIdx.x * 8 + wave) * 32;
    const int n0 = blockIdx.y * 64;
    v8f acc[2][4] = {};
    gemm_tile<KFPAD>(A, Bt, M, m0, n0, lane, acc);

    const int half = lane >> 4, l15 = lane & 15;
    #pragma unroll
    for (int nt = 0; nt < 4; ++nt) {
        int n = n0 + nt * 16 + l15;
        float bv = bias[n];
        #pragma unroll
        for (int t = 0; t < 2; ++t) {
            #pragma unroll
            for (int v = 0; v < 8; ++v) {
                int m = m0 + t * 16 + v + 8 * half;  // C layout: VGPR v, halves m / m+8
                float h = acc[t][nt][v] + bv;
                h *= h;
                if (m < M) C[(size_t)m * N + n] = __float2bfloat16(h);
            }
        }
    }
}

// ---------------------------------------------------------------------------
// Kernel 4: GEMM layer 2 fused with GEMV (W3):
//   h = (A @ W2 + b2)^2 ; partial[m, blockIdx.y] = sum_n(tile) h[m,n]*W3[n]
// Deterministic: each (row, blockIdx.y) slot written by exactly one wave.
// ---------------------------------------------------------------------------
__global__ __launch_bounds__(256)
void gemm_sq_dot_kernel(const __hip_bfloat16* __restrict__ A,
                        const __hip_bfloat16* __restrict__ Bt,
                        const float* __restrict__ bias,
                        const float* __restrict__ W3,
                        float* __restrict__ partial,
                        int M) {
    const int wave = threadIdx.x >> 5, lane = threadIdx.x & 31;
    const int m0 = (blockIdx.x * 8 + wave) * 32;
    const int n0 = blockIdx.y * 64;
    v8f acc[2][4] = {};
    gemm_tile<HID>(A, Bt, M, m0, n0, lane, acc);

    const int half = lane >> 4, l15 = lane & 15;
    #pragma unroll
    for (int t = 0; t < 2; ++t) {
        #pragma unroll
        for (int v = 0; v < 8; ++v) {
            float s = 0.f;
            #pragma unroll
            for (int nt = 0; nt < 4; ++nt) {
                int n = n0 + nt * 16 + l15;
                float h = acc[t][nt][v] + bias[n];
                h *= h;
                s += h * W3[n];
            }
            // reduce across the 16 lanes of each half (n dimension)
            #pragma unroll
            for (int off = 1; off < 16; off <<= 1) s += __shfl_xor(s, off, 16);
            if (l15 == 0) {
                int m = m0 + t * 16 + v + 8 * half;
                if (m < M) partial[(size_t)m * PW + blockIdx.y] = s;
            }
        }
    }
}

// ---------------------------------------------------------------------------
// Kernel 5: final reduce + log|.| + tower difference + clip
// ---------------------------------------------------------------------------
__global__ __launch_bounds__(256)
void final_kernel(const float* __restrict__ pa, const float* __restrict__ pb,
                  const float* __restrict__ b3a, const float* __restrict__ b3b,
                  const float* __restrict__ Wfa, const float* __restrict__ Wfb,
                  float* __restrict__ out, int B) {
    int m = blockIdx.x * blockDim.x + threadIdx.x;
    if (m >= B) return;
    float sa = 0.f, sb = 0.f;
    #pragma unroll
    for (int j = 0; j < PW; ++j) {
        sa += pa[(size_t)m * PW + j];
        sb += pb[(size_t)m * PW + j];
    }
    float ha = sa + b3a[0]; ha *= ha;
    float hb = sb + b3b[0]; hb *= hb;
    float o = logf(fabsf(ha)) * Wfa[0] - logf(fabsf(hb)) * Wfb[0];
    o = fminf(fmaxf(o, -1.0e6f), 1.0e6f);
    out[m] = o;
}

// ---------------------------------------------------------------------------
extern "C" void kernel_launch(void* const* d_in, const int* in_sizes, int n_in,
                              void* d_out, int out_size, void* d_ws, size_t ws_size,
                              hipStream_t stream) {
    const float* x   = (const float*)d_in[0];
    const float* W1a = (const float*)d_in[1];
    const float* b1a = (const float*)d_in[2];
    const float* W2a = (const float*)d_in[3];
    const float* b2a = (const float*)d_in[4];
    const float* W3a = (const float*)d_in[5];
    const float* b3a = (const float*)d_in[6];
    const float* Wfa = (const float*)d_in[7];
    const float* W1b = (const float*)d_in[8];
    const float* b1b = (const float*)d_in[9];
    const float* W2b = (const float*)d_in[10];
    const float* b2b = (const float*)d_in[11];
    const float* W3b = (const float*)d_in[12];
    const float* b3b = (const float*)d_in[13];
    const float* Wfb = (const float*)d_in[14];

    const int B = in_sizes[0] / 18;

    // workspace carve-up (256B aligned)
    char* w = (char*)d_ws;
    auto take = [&](size_t bytes) -> char* {
        char* p = w;
        w += (bytes + 255) & ~(size_t)255;
        return p;
    };
    __hip_bfloat16* feats = (__hip_bfloat16*)take((size_t)B * KFPAD * 2);
    __hip_bfloat16* h1a   = (__hip_bfloat16*)take((size_t)B * HID * 2);
    __hip_bfloat16* h1b   = (__hip_bfloat16*)take((size_t)B * HID * 2);
    __hip_bfloat16* W1at  = (__hip_bfloat16*)take((size_t)HID * KFPAD * 2);
    __hip_bfloat16* W1bt  = (__hip_bfloat16*)take((size_t)HID * KFPAD * 2);
    __hip_bfloat16* W2at  = (__hip_bfloat16*)take((size_t)HID * HID * 2);
    __hip_bfloat16* W2bt  = (__hip_bfloat16*)take((size_t)HID * HID * 2);
    float* pa = (float*)take((size_t)B * PW * 4);
    float* pb = (float*)take((size_t)B * PW * 4);

    // 1) features
    feats_kernel<<<(B + 7) / 8, 256, 0, stream>>>(x, feats, B);

    // 2) weight prep (bf16, transposed, K padded)
    int nW1 = KFPAD * HID, nW2 = HID * HID;
    wprep_kernel<<<(nW1 + 255) / 256, 256, 0, stream>>>(W1a, W1at, 729, HID, KFPAD);
    wprep_kernel<<<(nW1 + 255) / 256, 256, 0, stream>>>(W1b, W1bt, 729, HID, KFPAD);
    wprep_kernel<<<(nW2 + 255) / 256, 256, 0, stream>>>(W2a, W2at, HID, HID, HID);
    wprep_kernel<<<(nW2 + 255) / 256, 256, 0, stream>>>(W2b, W2bt, HID, HID, HID);

    // 3) layer 1 GEMMs (both towers)
    dim3 g1((B + 255) / 256, HID / 64);
    gemm_sq_kernel<<<g1, 256, 0, stream>>>(feats, W1at, b1a, h1a, B, HID);
    gemm_sq_kernel<<<g1, 256, 0, stream>>>(feats, W1bt, b1b, h1b, B, HID);

    // 4) layer 2 GEMMs fused with W3 GEMV (both towers)
    gemm_sq_dot_kernel<<<g1, 256, 0, stream>>>(h1a, W2at, b2a, W3a, pa, B);
    gemm_sq_dot_kernel<<<g1, 256, 0, stream>>>(h1b, W2bt, b2b, W3b, pb, B);

    // 5) final
    final_kernel<<<(B + 255) / 256, 256, 0, stream>>>(pa, pb, b3a, b3b, Wfa, Wfb,
                                                      (float*)d_out, B);
    (void)n_in; (void)out_size; (void)ws_size;
}
